// NanEchoBlock_69836168233476
// MI455X (gfx1250) — matrix operand order repaired and checked by reading, not compiled, not run on hardware
//
#include <hip/hip_runtime.h>
#include <hip/hip_bf16.h>
#include <math.h>

#define B_ 4
#define T_ 1024
#define C_ 768
#define H_ 12
#define HD_ 64

typedef __bf16 bf16_t;
typedef __attribute__((ext_vector_type(16))) __bf16 v16bf;
typedef __attribute__((ext_vector_type(8)))  __bf16 v8bf;
typedef __attribute__((ext_vector_type(8)))  float  v8f;

union FragBF { v16bf v; v8bf h[2]; };

__device__ __forceinline__ v8f wmma_bf16(v16bf a, v16bf b, v8f c) {
  // D = A(16x32 bf16) x B(32x16 bf16) + C(16x16 f32)
  return __builtin_amdgcn_wmma_f32_16x16x32_bf16(false, a, false, b, (short)0, c,
                                                 false, false);
}

// A-matrix fragment: lane L (L<16) holds row M=L, K = {k..k+7, k+16..k+23};
// lane L+16 holds row M=L, K = {k+8..k+15, k+24..k+31}. (ISA 7.12.2)
__device__ __forceinline__ v16bf load_frag_a(const bf16_t* tile, int ld, int lane) {
  FragBF f;
  int m  = lane & 15;
  int hk = (lane >> 4) << 3;
  const bf16_t* p = tile + (size_t)m * ld + hk;
  f.h[0] = *(const v8bf*)(p);
  f.h[1] = *(const v8bf*)(p + 16);
  return f.v;
}

// B-matrix fragment from row-major W[N,K] (so B column n = W row n):
// lane L holds N=L&15, K = {8*(L>>4)..+7, 16+8*(L>>4)..+7} (same order as A).
__device__ __forceinline__ v16bf load_frag_b(const bf16_t* tile, int ld, int lane) {
  FragBF f;
  int n  = lane & 15;
  int ko = (lane >> 4) << 3;
  const bf16_t* p = tile + (size_t)n * ld + ko;
  f.h[0] = *(const v8bf*)(p);
  f.h[1] = *(const v8bf*)(p + 16);
  return f.v;
}

// ---------------- elementwise converts ----------------
__global__ void f32_to_bf16_kernel(const float* __restrict__ in,
                                   bf16_t* __restrict__ out, int n) {
  int i = blockIdx.x * blockDim.x + threadIdx.x;
  if (i < n) out[i] = (bf16_t)in[i];
}

__global__ void mask_mul_bf16_kernel(const float* __restrict__ w,
                                     const float* __restrict__ m,
                                     bf16_t* __restrict__ out, int n) {
  int i = blockIdx.x * blockDim.x + threadIdx.x;
  if (i < n) out[i] = (bf16_t)(w[i] * m[i]);
}

// ---------------- layernorm: one wave per row of 768 ----------------
__global__ __launch_bounds__(256) void ln_kernel(const float* __restrict__ x,
                                                 const float* __restrict__ g,
                                                 const float* __restrict__ beta,
                                                 bf16_t* __restrict__ out_bf,
                                                 float* __restrict__ out_f) {
  int lane = threadIdx.x & 31;
  int wave = threadIdx.x >> 5;
  size_t row = (size_t)blockIdx.x * 8 + wave;
  const float* xr = x + row * C_;
  float v[24];
  float s = 0.f;
#pragma unroll
  for (int i = 0; i < 24; ++i) { v[i] = xr[lane + 32 * i]; s += v[i]; }
#pragma unroll
  for (int m = 16; m >= 1; m >>= 1) s += __shfl_xor(s, m, 32);
  float mu = s * (1.0f / C_);
  float q = 0.f;
#pragma unroll
  for (int i = 0; i < 24; ++i) { float d = v[i] - mu; q += d * d; }
#pragma unroll
  for (int m = 16; m >= 1; m >>= 1) q += __shfl_xor(q, m, 32);
  float inv = rsqrtf(q * (1.0f / C_) + 1e-5f);
#pragma unroll
  for (int i = 0; i < 24; ++i) {
    int c = lane + 32 * i;
    float hv = (v[i] - mu) * inv * g[c] + beta[c];
    out_bf[row * C_ + c] = (bf16_t)hv;
    if (out_f) out_f[row * C_ + c] = hv;
  }
}

// ---------------- adaptive threshold (tiny, one block per batch) ----------------
__global__ __launch_bounds__(256) void thr_kernel(const float* __restrict__ hf,
                                                  const float* __restrict__ w1,
                                                  const float* __restrict__ b1,
                                                  const float* __restrict__ w2,
                                                  const float* __restrict__ b2,
                                                  const float* __restrict__ cw,
                                                  const float* __restrict__ cb,
                                                  float* __restrict__ thr_out) {
  __shared__ float xm[C_];
  __shared__ float hid[C_ / 2];
  int b = blockIdx.x, tid = threadIdx.x;
  const float* base = hf + (size_t)b * T_ * C_;
  float s0 = 0.f, s1 = 0.f, s2 = 0.f;
  for (int t = 0; t < T_; ++t) {
    const float* r = base + (size_t)t * C_;
    s0 += r[tid]; s1 += r[tid + 256]; s2 += r[tid + 512];
  }
  xm[tid] = s0 * (1.0f / T_);
  xm[tid + 256] = s1 * (1.0f / T_);
  xm[tid + 512] = s2 * (1.0f / T_);
  __syncthreads();
  for (int j = tid; j < C_ / 2; j += 256) {
    float a = b1[j];
    for (int c = 0; c < C_; ++c) a += xm[c] * w1[(size_t)j * C_ + c];
    hid[j] = fmaxf(a, 0.0f);
  }
  __syncthreads();
  if (tid == 0) {
    float a = b2[0];
    for (int j = 0; j < C_ / 2; ++j) a += hid[j] * w2[j];
    float base_s = 1.0f / (1.0f + __expf(-a));
    float cg = cb[0];
    for (int c = 0; c < C_; ++c) cg += xm[c] * cw[c];
    cg = 1.0f / (1.0f + __expf(-cg));
    float th = 0.3f + 0.6f * base_s;
    th = th * (1.0f + cg * 0.5f);
    thr_out[b] = fminf(fmaxf(th, 0.3f), 0.9f);
  }
}

// ---------------- WMMA GEMM: out[M,N] = A[M,K] @ W[N,K]^T (+bias, epilogue) ----
// EPI 0: bias -> bf16; EPI 1: bias + residual -> f32; EPI 2: bias+GELU -> bf16;
// EPI 3: bias, then scale the K-projection block (cols [C,2C)) by 1/sqrt(HD) -> bf16
template <int EPI>
__global__ __launch_bounds__(256) void gemm_kernel(const bf16_t* __restrict__ A,
                                                   const bf16_t* __restrict__ W,
                                                   const float* __restrict__ bias,
                                                   void* __restrict__ outp,
                                                   const float* __restrict__ res,
                                                   int M, int N, int K) {
  int lane = threadIdx.x & 31;
  int wave = threadIdx.x >> 5;
  int wr = wave >> 2, wc = wave & 3;           // 2x4 wave grid
  int bm = blockIdx.y * 128 + wr * 64;         // wave: 64 rows
  int bn = blockIdx.x * 128 + wc * 32;         // wave: 32 cols
  v8f acc[4][2] = {};
#pragma unroll 2
  for (int k = 0; k < K; k += 32) {
    v16bf a[4], bw[2];
#pragma unroll
    for (int mt = 0; mt < 4; ++mt) {
      const bf16_t* pa = A + (size_t)(bm + mt * 16) * K + k;
      a[mt] = load_frag_a(pa, K, lane);
      __builtin_prefetch(pa + 64, 0, 0);  // two K-tiles ahead -> global_prefetch
    }
#pragma unroll
    for (int nt = 0; nt < 2; ++nt) {
      const bf16_t* pb = W + (size_t)(bn + nt * 16) * K + k;
      bw[nt] = load_frag_b(pb, K, lane);
      __builtin_prefetch(pb + 64, 0, 0);
    }
#pragma unroll
    for (int mt = 0; mt < 4; ++mt)
#pragma unroll
      for (int nt = 0; nt < 2; ++nt)
        acc[mt][nt] = wmma_bf16(a[mt], bw[nt], acc[mt][nt]);
  }
  int rbase = (lane >> 4) << 3;
  int ncol = lane & 15;
#pragma unroll
  for (int mt = 0; mt < 4; ++mt) {
#pragma unroll
    for (int nt = 0; nt < 2; ++nt) {
      int col = bn + nt * 16 + ncol;
      float bv = bias[col];
#pragma unroll
      for (int r = 0; r < 8; ++r) {
        int row = bm + mt * 16 + rbase + r;
        float v = acc[mt][nt][r] + bv;
        size_t idx = (size_t)row * N + col;
        if constexpr (EPI == 0) {
          ((bf16_t*)outp)[idx] = (bf16_t)v;
        } else if constexpr (EPI == 1) {
          ((float*)outp)[idx] = v + res[idx];
        } else if constexpr (EPI == 2) {
          float gl = 0.5f * v * (1.0f + erff(v * 0.70710678118f));
          ((bf16_t*)outp)[idx] = (bf16_t)gl;
        } else {
          // fused QKV epilogue: pre-scale K block so attention skips the
          // 1/sqrt(HD) multiply on every score element
          float vv = (col >= C_ && col < 2 * C_) ? v * 0.125f : v;
          ((bf16_t*)outp)[idx] = (bf16_t)vv;
        }
      }
    }
  }
}

// ---------------- flash attention with shifted-softplus threshold ----------------
// block: 128 threads (4 waves); each wave: 16 query rows; key tiles of 32.
// Key algebraic trick: softmax(softplus(s-t)-softplus(-t-s)) only needs
//   r = e^{f(s)} = (1+e^{s-t}) / (1+e^{-s-t}),
// so the online softmax runs in ratio space (track running row-max of r),
// eliminating all v_log chains and one exp per element.
__global__ __launch_bounds__(128) void attn_kernel(const bf16_t* __restrict__ qkv,
                                                   const float* __restrict__ thr,
                                                   bf16_t* __restrict__ out) {
  __shared__ __align__(16) bf16_t VT[64 * 40];     // V tile transposed, padded ld=40
  __shared__ __align__(16) bf16_t P[4][16 * 32];   // per-wave probability tile
  const int ld = 3 * C_;
  int lane = threadIdx.x & 31;
  int wave = threadIdx.x >> 5;
  int qt = blockIdx.x, hh = blockIdx.y, b = blockIdx.z;
  int qrow0 = qt * 64 + wave * 16;
  const bf16_t* qb  = qkv + (size_t)b * T_ * ld + hh * HD_;
  const bf16_t* kbp = qb + C_;
  const bf16_t* vbp = qb + 2 * C_;
  v16bf qf0 = load_frag_a(qb + (size_t)qrow0 * ld, ld, lane);
  v16bf qf1 = load_frag_a(qb + (size_t)qrow0 * ld + 32, ld, lane);
  float tb = thr[b];
  float mrow[8], lrow[8];   // mrow = running max of r (ratio space)
#pragma unroll
  for (int r = 0; r < 8; ++r) { mrow[r] = 0.f; lrow[r] = 0.f; }
  v8f o[4] = {};
  int rbase = (lane >> 4) << 3;
  int ncol = lane & 15;
  int kend = qt * 64 + 64;  // causal: no keys beyond last query in this block
  for (int kb0 = 0; kb0 < kend; kb0 += 32) {
    __syncthreads();
    {  // cooperatively stage V^T tile (32 keys x 64 dims -> VT[d][key])
      int t = threadIdx.x;
      int key = kb0 + (t >> 2);
      int d0 = (t & 3) * 16;
      const bf16_t* vp = vbp + (size_t)key * ld + d0;
      v8bf v0 = *(const v8bf*)vp;
      v8bf v1 = *(const v8bf*)(vp + 8);
#pragma unroll
      for (int i = 0; i < 8; ++i) {
        VT[(d0 + i) * 40 + (key - kb0)] = v0[i];
        VT[(d0 + 8 + i) * 40 + (key - kb0)] = v1[i];
      }
    }
    __syncthreads();
    // S = Q @ K^T for 16 rows x 32 keys (K already pre-scaled by 1/sqrt(HD))
    v8f s[2] = {};
#pragma unroll
    for (int c = 0; c < 2; ++c) {
      v16bf k0 = load_frag_b(kbp + (size_t)(kb0 + c * 16) * ld, ld, lane);
      v16bf k1 = load_frag_b(kbp + (size_t)(kb0 + c * 16) * ld + 32, ld, lane);
      s[c] = wmma_bf16(qf0, k0, s[c]);
      s[c] = wmma_bf16(qf1, k1, s[c]);
    }
    // threshold transform in ratio space; causal mask only on boundary tiles
    float rv[2][8];
    bool tile_masked = (kb0 + 32 > qrow0);  // wave-uniform
#pragma unroll
    for (int c = 0; c < 2; ++c)
#pragma unroll
      for (int r = 0; r < 8; ++r) {
        float sv = s[c][r];
        if (tile_masked) {
          int qi = qrow0 + rbase + r;
          int ki = kb0 + c * 16 + ncol;
          if (ki > qi) sv = -1.0e9f;  // -> r = 0
        }
        float ea = __expf(sv - tb);
        float eb = __expf(-sv - tb);
        rv[c][r] = (1.0f + ea) * __builtin_amdgcn_rcpf(1.0f + eb);
      }
    // online softmax in ratio space (row m lives in one 16-lane half, vgpr m&7)
#pragma unroll
    for (int r = 0; r < 8; ++r) {
      float rm = fmaxf(rv[0][r], rv[1][r]);
#pragma unroll
      for (int m = 8; m >= 1; m >>= 1) rm = fmaxf(rm, __shfl_xor(rm, m, 32));
      float mn = fmaxf(mrow[r], rm);
      float rinv = __builtin_amdgcn_rcpf(mn);
      float al = mrow[r] * rinv;   // r_max_old / r_max_new
      mrow[r] = mn;
      float ps = 0.f;
#pragma unroll
      for (int c = 0; c < 2; ++c) {
        float p = rv[c][r] * rinv;
        rv[c][r] = p;
        ps += p;
      }
#pragma unroll
      for (int m = 8; m >= 1; m >>= 1) ps += __shfl_xor(ps, m, 32);
      lrow[r] = lrow[r] * al + ps;
#pragma unroll
      for (int t = 0; t < 4; ++t) o[t][r] *= al;
    }
    // P through LDS: C-layout -> A-layout re-fragmentation
    bf16_t* pw = &P[wave][0];
#pragma unroll
    for (int c = 0; c < 2; ++c)
#pragma unroll
      for (int r = 0; r < 8; ++r)
        pw[(rbase + r) * 32 + c * 16 + ncol] = (bf16_t)rv[c][r];
    v16bf pf = load_frag_a(pw, 32, lane);
    // O += P @ V
#pragma unroll
    for (int t = 0; t < 4; ++t) {
      v16bf vf = load_frag_b(&VT[t * 16 * 40], 40, lane);
      o[t] = wmma_bf16(pf, vf, o[t]);
    }
  }
  // normalize + write head-merged output as bf16 for the out-projection GEMM
#pragma unroll
  for (int r = 0; r < 8; ++r) {
    float li = __builtin_amdgcn_rcpf(lrow[r]);
    int qi = qrow0 + rbase + r;
#pragma unroll
    for (int t = 0; t < 4; ++t) {
      int d = t * 16 + ncol;
      out[((size_t)b * T_ + qi) * C_ + hh * HD_ + d] = (bf16_t)(o[t][r] * li);
    }
  }
}

extern "C" void kernel_launch(void* const* d_in, const int* in_sizes, int n_in,
                              void* d_out, int out_size, void* d_ws, size_t ws_size,
                              hipStream_t stream) {
  (void)in_sizes; (void)n_in; (void)out_size; (void)ws_size;
  const float* x      = (const float*)d_in[0];
  const float* ln1_g  = (const float*)d_in[1];
  const float* ln1_b  = (const float*)d_in[2];
  const float* qkv_w  = (const float*)d_in[3];
  const float* qkv_b  = (const float*)d_in[4];
  const float* out_w  = (const float*)d_in[5];
  const float* out_b  = (const float*)d_in[6];
  const float* cmask  = (const float*)d_in[7];
  const float* thr_w1 = (const float*)d_in[8];
  const float* thr_b1 = (const float*)d_in[9];
  const float* thr_w2 = (const float*)d_in[10];
  const float* thr_b2 = (const float*)d_in[11];
  const float* cog_w  = (const float*)d_in[12];
  const float* cog_b  = (const float*)d_in[13];
  const float* ln2_g  = (const float*)d_in[14];
  const float* ln2_b  = (const float*)d_in[15];
  const float* mlp_w1 = (const float*)d_in[16];
  const float* mlp_b1 = (const float*)d_in[17];
  const float* mlp_w2 = (const float*)d_in[18];
  const float* mlp_b2 = (const float*)d_in[19];

  char* ws = (char*)d_ws;
  size_t off = 0;
  auto alloc = [&](size_t bytes) -> void* {
    void* p = ws + off;
    off = (off + bytes + 255) & ~(size_t)255;
    return p;
  };
  const int M = B_ * T_;  // 4096
  bf16_t* h_bf  = (bf16_t*)alloc((size_t)M * C_ * 2);
  float*  h_f32 = (float*) alloc((size_t)M * C_ * 4);
  float*  thr   = (float*) alloc(B_ * sizeof(float));
  bf16_t* w_qkv = (bf16_t*)alloc((size_t)3 * C_ * C_ * 2);
  bf16_t* w_out = (bf16_t*)alloc((size_t)C_ * C_ * 2);
  bf16_t* w_m1  = (bf16_t*)alloc((size_t)4 * C_ * C_ * 2);
  bf16_t* w_m2  = (bf16_t*)alloc((size_t)4 * C_ * C_ * 2);
  bf16_t* qkv   = (bf16_t*)alloc((size_t)M * 3 * C_ * 2);
  bf16_t* att_o = (bf16_t*)alloc((size_t)M * C_ * 2);
  float*  x2    = (float*) alloc((size_t)M * C_ * 4);
  bf16_t* h2_bf = (bf16_t*)alloc((size_t)M * C_ * 2);
  bf16_t* m1    = (bf16_t*)alloc((size_t)M * 4 * C_ * 2);

  int n;
  n = 3 * C_ * C_;
  f32_to_bf16_kernel<<<(n + 255) / 256, 256, 0, stream>>>(qkv_w, w_qkv, n);
  n = C_ * C_;
  mask_mul_bf16_kernel<<<(n + 255) / 256, 256, 0, stream>>>(out_w, cmask, w_out, n);
  n = 4 * C_ * C_;
  f32_to_bf16_kernel<<<(n + 255) / 256, 256, 0, stream>>>(mlp_w1, w_m1, n);
  f32_to_bf16_kernel<<<(n + 255) / 256, 256, 0, stream>>>(mlp_w2, w_m2, n);

  ln_kernel<<<M / 8, 256, 0, stream>>>(x, ln1_g, ln1_b, h_bf, h_f32);
  thr_kernel<<<B_, 256, 0, stream>>>(h_f32, thr_w1, thr_b1, thr_w2, thr_b2,
                                     cog_w, cog_b, thr);
  gemm_kernel<3><<<dim3(3 * C_ / 128, M / 128), 256, 0, stream>>>(
      h_bf, w_qkv, qkv_b, qkv, nullptr, M, 3 * C_, C_);
  attn_kernel<<<dim3(T_ / 64, H_, B_), 128, 0, stream>>>(qkv, thr, att_o);
  gemm_kernel<1><<<dim3(C_ / 128, M / 128), 256, 0, stream>>>(
      att_o, w_out, out_b, x2, x, M, C_, C_);
  ln_kernel<<<M / 8, 256, 0, stream>>>(x2, ln2_g, ln2_b, h2_bf, nullptr);
  gemm_kernel<2><<<dim3(4 * C_ / 128, M / 128), 256, 0, stream>>>(
      h2_bf, w_m1, mlp_b1, m1, nullptr, M, 4 * C_, C_);
  gemm_kernel<1><<<dim3(C_ / 128, M / 128), 256, 0, stream>>>(
      m1, w_m2, mlp_b2, (float*)d_out, x2, M, C_, 4 * C_);
}
